// Autoformer_70317204570383
// MI455X (gfx1250) — matrix operand order repaired and checked by reading, not compiled, not run on hardware
//
#include <hip/hip_runtime.h>
#include <hip/hip_bf16.h>

// ---------------- types for WMMA ----------------
typedef __attribute__((ext_vector_type(16))) _Float16 v16h;
typedef __attribute__((ext_vector_type(8)))  _Float16 v8h;
typedef __attribute__((ext_vector_type(8)))  float    v8f;

// ---------------- static model config ----------------
#define B_    16
#define L_    1024
#define C_    7
#define D_    512
#define DFF_  2048
#define TOPK_ 6
#define BL_   (B_ * L_)
#define BLD_  (BL_ * D_)            // 8,388,608
#define BLC_  (BL_ * C_)            // 114,688

// GEMM epilogue flags
#define GF_BIAS 1
#define GF_GELU 2
#define GF_RES  4
#define GF_F32  8
#define GF_F16  16

// =====================================================================
// WMMA GEMM: C[M,N] = A[M,K](f16) x Bt[N,K](f16, pre-transposed) [+bias]
//            [+residual] [gelu] -> f32 C and/or f16 Ch
// One wave -> 16(M) x 64(N) tile (4 accumulators). block=128 (4 waves).
// grid = (N/256, M/16). Requires K%32==0, N%256==0, M%16==0.
// A and B fragments both use contiguous k = 16*hi + s (wave32).
// =====================================================================
__global__ __launch_bounds__(128)
void k_wmma_gemm(const _Float16* __restrict__ A, const _Float16* __restrict__ Bt,
                 const float* __restrict__ bias, const float* __restrict__ R,
                 float* __restrict__ C, _Float16* __restrict__ Ch,
                 int M, int N, int K, int flags) {
  const int lane   = threadIdx.x & 31;
  const int wave   = threadIdx.x >> 5;
  const int m0     = blockIdx.y * 16;
  const int n0     = blockIdx.x * 256 + wave * 64;
  const int hi     = (lane < 16) ? 0 : 1;     // lane half
  const int lr     = lane & 15;

  v8f acc[4] = {v8f{}, v8f{}, v8f{}, v8f{}};

  const _Float16* ap = A + (size_t)(m0 + lr) * K + hi * 16;
  const _Float16* bp[4];
#pragma unroll
  for (int t = 0; t < 4; ++t)
    bp[t] = Bt + (size_t)(n0 + t * 16 + lr) * K + hi * 16;

  for (int k0 = 0; k0 < K; k0 += 32) {
    if (k0 + 128 < K) {
      __builtin_prefetch((const void*)(ap + k0 + 128), 0, 1);
      __builtin_prefetch((const void*)(bp[0] + k0 + 128), 0, 1);
    }
    // A fragment: lane holds row m0+lr, 16 consecutive K at k0+16*hi.
    v8h alo = *(const v8h*)(ap + k0);
    v8h ahi = *(const v8h*)(ap + k0 + 8);
    v16h a;
#pragma unroll
    for (int e = 0; e < 8; ++e) { a[e] = alo[e]; a[e + 8] = ahi[e]; }
#pragma unroll
    for (int t = 0; t < 4; ++t) {
      // B fragment: lane holds col n0+t*16+lr, 16 consecutive K at k0+16*hi.
      v16h b = *(const v16h*)(bp[t] + k0);
      acc[t] = __builtin_amdgcn_wmma_f32_16x16x32_f16(
          false, a, false, b, (short)0, acc[t], false, false);
    }
  }

  // Epilogue: acc[t][r] -> row m0 + r + 8*hi, col n0 + t*16 + lr
#pragma unroll
  for (int t = 0; t < 4; ++t) {
    const int col = n0 + t * 16 + lr;
    const float bv = (flags & GF_BIAS) ? bias[col] : 0.0f;
#pragma unroll
    for (int r = 0; r < 8; ++r) {
      const int row = m0 + r + hi * 8;
      float v = acc[t][r] + bv;
      if (flags & GF_RES) v += R[(size_t)row * N + col];
      if (flags & GF_GELU) {
        float x = v;
        v = 0.5f * x * (1.0f + tanhf(0.7978845608028654f * (x + 0.044715f * x * x * x)));
      }
      if (flags & GF_F32) C[(size_t)row * N + col] = v;
      if (flags & GF_F16) Ch[(size_t)row * N + col] = (_Float16)v;
    }
  }
}

// ------------- conversions -------------
__global__ void k_f32_to_f16(const float* __restrict__ x, _Float16* __restrict__ y, int n) {
  int i = blockIdx.x * blockDim.x + threadIdx.x;
  if (i < n) y[i] = (_Float16)x[i];
}
// W (K,N) f32 -> Wt (N,K) f16
__global__ void k_transpose_f16(const float* __restrict__ w, _Float16* __restrict__ wt,
                                int K, int N) {
  int i = blockIdx.x * blockDim.x + threadIdx.x;
  if (i >= K * N) return;
  int n = i % N, k = i / N;
  wt[(size_t)n * K + k] = (_Float16)w[i];
}

// ------------- series decomposition (k=25, edge-replicated mavg) -------------
// seasonal = x - trend (optional); trend written or accumulated (optional).
__global__ void k_series_decomp(const float* __restrict__ x, float* __restrict__ seasonal,
                                float* __restrict__ trend, int Ch, int n, int accum) {
  int i = blockIdx.x * blockDim.x + threadIdx.x;
  if (i >= n) return;
  int c = i % Ch; int l = (i / Ch) % L_; int b = i / (Ch * L_);
  const float* xb = x + (size_t)b * L_ * Ch;
  float s = 0.0f;
  for (int j = -12; j <= 12; ++j) {
    int ll = l + j; ll = ll < 0 ? 0 : (ll >= L_ ? L_ - 1 : ll);
    s += xb[(size_t)ll * Ch + c];
  }
  float t = s * (1.0f / 25.0f);
  if (seasonal) seasonal[i] = x[i] - t;
  if (trend) { if (accum) trend[i] += t; else trend[i] = t; }
}

// ------------- circular conv1d, K=3, weight (3,Cin,Cout) -------------
__global__ void k_circ_conv3(const float* __restrict__ x, const float* __restrict__ w,
                             float* __restrict__ y, int Cin, int Cout, int accum, int n) {
  int i = blockIdx.x * blockDim.x + threadIdx.x;
  if (i >= n) return;
  int co = i % Cout; int l = (i / Cout) % L_; int b = i / (Cout * L_);
  const float* xb = x + (size_t)b * L_ * Cin;
  float acc = 0.0f;
  for (int t = 0; t < 3; ++t) {
    int ll = l + t - 1; ll = (ll + L_) % L_;
    const float* xr = xb + (size_t)ll * Cin;
    const float* wr = w + (size_t)t * Cin * Cout + co;
    for (int ci = 0; ci < Cin; ++ci) acc += xr[ci] * wr[(size_t)ci * Cout];
  }
  if (accum) y[i] += acc; else y[i] = acc;
}

// ------------- mean autocorrelation: corr[b,tau] = (1/D) sum_{t,d} q[(t+tau)%L,d]*k[t,d]
__global__ void k_mean_corr(const float* __restrict__ Q, const float* __restrict__ Kv,
                            float* __restrict__ corr) {
  int tau = blockIdx.x, b = blockIdx.y;
  const float* qb = Q + (size_t)b * L_ * D_;
  const float* kb = Kv + (size_t)b * L_ * D_;
  float p = 0.0f;
  for (int idx = threadIdx.x; idx < L_ * D_; idx += 256) {
    int t = idx >> 9;                 // D_ = 512
    int tq = t + tau; if (tq >= L_) tq -= L_;
    p += qb[(size_t)tq * D_ + (idx & (D_ - 1))] * kb[idx];
  }
  __shared__ float red[256];
  red[threadIdx.x] = p; __syncthreads();
  for (int s = 128; s > 0; s >>= 1) {
    if (threadIdx.x < s) red[threadIdx.x] += red[threadIdx.x + s];
    __syncthreads();
  }
  if (threadIdx.x == 0) corr[b * L_ + tau] = red[0] * (1.0f / (float)D_);
}

// ------------- top-6 + softmax per batch -------------
__global__ void k_topk_softmax(const float* __restrict__ corr, float* __restrict__ wts,
                               int* __restrict__ delays) {
  int b = blockIdx.x;
  __shared__ float vals[L_];
  __shared__ float rv[256]; __shared__ int ri[256];
  __shared__ float topv[TOPK_]; __shared__ int topi[TOPK_];
  for (int i = threadIdx.x; i < L_; i += 256) vals[i] = corr[b * L_ + i];
  __syncthreads();
  for (int kk = 0; kk < TOPK_; ++kk) {
    float bv = -3.0e38f; int bi = 0;
    for (int i = threadIdx.x; i < L_; i += 256) {
      float v = vals[i]; if (v > bv) { bv = v; bi = i; }
    }
    rv[threadIdx.x] = bv; ri[threadIdx.x] = bi; __syncthreads();
    for (int s = 128; s > 0; s >>= 1) {
      if (threadIdx.x < s && rv[threadIdx.x + s] > rv[threadIdx.x]) {
        rv[threadIdx.x] = rv[threadIdx.x + s]; ri[threadIdx.x] = ri[threadIdx.x + s];
      }
      __syncthreads();
    }
    if (threadIdx.x == 0) { topv[kk] = rv[0]; topi[kk] = ri[0]; vals[ri[0]] = -3.0e38f; }
    __syncthreads();
  }
  if (threadIdx.x == 0) {
    float mx = topv[0], s = 0.0f, e[TOPK_];
    for (int j = 0; j < TOPK_; ++j) { e[j] = expf(topv[j] - mx); s += e[j]; }
    for (int j = 0; j < TOPK_; ++j) { wts[b * TOPK_ + j] = e[j] / s; delays[b * TOPK_ + j] = topi[j]; }
  }
}

// ------------- time-delay aggregation -------------
__global__ void k_time_delay_agg(const float* __restrict__ V, const float* __restrict__ wts,
                                 const int* __restrict__ delays, float* __restrict__ out, int n) {
  int i = blockIdx.x * blockDim.x + threadIdx.x;
  if (i >= n) return;
  int d = i % D_; int l = (i / D_) % L_; int b = i / (D_ * L_);
  const float* vb = V + (size_t)b * L_ * D_;
  float acc = 0.0f;
  for (int j = 0; j < TOPK_; ++j) {
    int idx = l + delays[b * TOPK_ + j]; if (idx >= L_) idx -= L_;
    acc += wts[b * TOPK_ + j] * vb[(size_t)idx * D_ + d];
  }
  out[i] = acc;
}

// ------------- my_layernorm: phase 1 (per-row LN over D) -------------
__global__ void k_ln_rows(const float* __restrict__ x, const float* __restrict__ g,
                          const float* __restrict__ bb, float* __restrict__ y, int rows) {
  int r = blockIdx.x * blockDim.x + threadIdx.x;
  if (r >= rows) return;
  const float* xr = x + (size_t)r * D_;
  float mu = 0.0f;
  for (int d = 0; d < D_; ++d) mu += xr[d];
  mu *= (1.0f / D_);
  float var = 0.0f;
  for (int d = 0; d < D_; ++d) { float t = xr[d] - mu; var += t * t; }
  var *= (1.0f / D_);
  float inv = rsqrtf(var + 1e-5f);
  float* yr = y + (size_t)r * D_;
  for (int d = 0; d < D_; ++d) yr[d] = (xr[d] - mu) * inv * g[d] + bb[d];
}
// phase 2: subtract per-(b,d) mean over L
__global__ void k_sub_seqmean(const float* __restrict__ x, float* __restrict__ y, int n) {
  int i = blockIdx.x * blockDim.x + threadIdx.x;
  if (i >= n) return;
  int d = i % D_; int b = i / D_;
  const float* xb = x + (size_t)b * L_ * D_ + d;
  float mu = 0.0f;
  for (int l = 0; l < L_; ++l) mu += xb[(size_t)l * D_];
  mu *= (1.0f / L_);
  float* yb = y + (size_t)b * L_ * D_ + d;
  for (int l = 0; l < L_; ++l) yb[(size_t)l * D_] = xb[(size_t)l * D_] - mu;
}

// ------------- small projection D->C (+bias +add) -------------
__global__ void k_proj_c(const float* __restrict__ x, const float* __restrict__ w,
                         const float* __restrict__ bias, const float* __restrict__ add,
                         float* __restrict__ y, int n) {
  int i = blockIdx.x * blockDim.x + threadIdx.x;
  if (i >= n) return;
  int c = i % C_; int r = i / C_;
  const float* xr = x + (size_t)r * D_;
  float acc = bias[c];
  for (int d = 0; d < D_; ++d) acc += xr[d] * w[(size_t)d * C_ + c];
  if (add) acc += add[i];
  y[i] = acc;
}

// ------------- final prediction: out[b] = dot(dec_out[b,:], pred_w) + pred_b
__global__ void k_pred(const float* __restrict__ x, const float* __restrict__ w,
                       const float* __restrict__ bias, float* __restrict__ out) {
  int b = blockIdx.x;
  const float* xb = x + (size_t)b * L_ * C_;
  float p = 0.0f;
  for (int i = threadIdx.x; i < L_ * C_; i += 256) p += xb[i] * w[i];
  __shared__ float red[256];
  red[threadIdx.x] = p; __syncthreads();
  for (int s = 128; s > 0; s >>= 1) {
    if (threadIdx.x < s) red[threadIdx.x] += red[threadIdx.x + s];
    __syncthreads();
  }
  if (threadIdx.x == 0) out[b] = red[0] + bias[0];
}

// =====================================================================
// Host-side orchestration (stream-only, graph-capture safe)
// =====================================================================
static inline dim3 g1(long n) { return dim3((unsigned)((n + 255) / 256)); }

static void gemm(hipStream_t st, const _Float16* A, const _Float16* Bt,
                 const float* bias, const float* R, float* C, _Float16* Ch,
                 int M, int N, int K, int flags) {
  k_wmma_gemm<<<dim3(N / 256, M / 16), dim3(128), 0, st>>>(A, Bt, bias, R, C, Ch, M, N, K, flags);
}

struct AttnW { const float *wq,*bq,*wk,*bk,*wv,*bv,*wo,*bo; };

static void run_attn(hipStream_t st, const float* xq, const float* xkv, const AttnW& aw,
                     const float* resid, float* out,
                     float* q, float* k, float* v,
                     float* corr, float* wts, int* delays,
                     _Float16* xh, _Float16* wt) {
  k_f32_to_f16<<<g1(BLD_), 256, 0, st>>>(xq, xh, BLD_);
  k_transpose_f16<<<g1(D_ * D_), 256, 0, st>>>(aw.wq, wt, D_, D_);
  gemm(st, xh, wt, aw.bq, nullptr, q, nullptr, BL_, D_, D_, GF_BIAS | GF_F32);
  if (xkv != xq) k_f32_to_f16<<<g1(BLD_), 256, 0, st>>>(xkv, xh, BLD_);
  k_transpose_f16<<<g1(D_ * D_), 256, 0, st>>>(aw.wk, wt, D_, D_);
  gemm(st, xh, wt, aw.bk, nullptr, k, nullptr, BL_, D_, D_, GF_BIAS | GF_F32);
  k_transpose_f16<<<g1(D_ * D_), 256, 0, st>>>(aw.wv, wt, D_, D_);
  gemm(st, xh, wt, aw.bv, nullptr, v, nullptr, BL_, D_, D_, GF_BIAS | GF_F32);
  k_mean_corr<<<dim3(L_, B_), 256, 0, st>>>(q, k, corr);
  k_topk_softmax<<<B_, 256, 0, st>>>(corr, wts, delays);
  k_time_delay_agg<<<g1(BLD_), 256, 0, st>>>(v, wts, delays, q, BLD_);  // agg -> q
  k_f32_to_f16<<<g1(BLD_), 256, 0, st>>>(q, xh, BLD_);
  k_transpose_f16<<<g1(D_ * D_), 256, 0, st>>>(aw.wo, wt, D_, D_);
  gemm(st, xh, wt, aw.bo, resid, out, nullptr, BL_, D_, D_, GF_BIAS | GF_RES | GF_F32);
}

// z = x + w2(gelu(w1(x))); FFN hidden chunked per batch (f16, 4MB)
static void run_ffn(hipStream_t st, const float* x, const float* w1, const float* w2,
                    float* z, _Float16* xh, _Float16* wtA, _Float16* wtB, _Float16* hh) {
  k_f32_to_f16<<<g1(BLD_), 256, 0, st>>>(x, xh, BLD_);
  k_transpose_f16<<<g1(D_ * DFF_), 256, 0, st>>>(w1, wtA, D_, DFF_);
  k_transpose_f16<<<g1(DFF_ * D_), 256, 0, st>>>(w2, wtB, DFF_, D_);
  for (int b = 0; b < B_; ++b) {
    size_t off = (size_t)b * L_ * D_;
    gemm(st, xh + off, wtA, nullptr, nullptr, nullptr, hh, L_, DFF_, D_, GF_GELU | GF_F16);
    gemm(st, hh, wtB, nullptr, x + off, z + off, nullptr, L_, D_, DFF_, GF_RES | GF_F32);
  }
}

extern "C" void kernel_launch(void* const* d_in, const int* in_sizes, int n_in,
                              void* d_out, int out_size, void* d_ws, size_t ws_size,
                              hipStream_t stream) {
  (void)in_sizes; (void)out_size; (void)ws_size;
  const float* P[50];
  for (int i = 0; i < 50; ++i) P[i] = (i < n_in) ? (const float*)d_in[i] : nullptr;

  // ---- workspace carving (bump allocator, 256B aligned) ----
  char* wp = (char*)d_ws;
  auto carve = [&](size_t bytes) -> void* {
    void* r = (void*)wp; wp += (bytes + 255) & ~(size_t)255; return r;
  };
  float* fb0 = (float*)carve((size_t)BLD_ * 4);   // enc state
  float* fb1 = (float*)carve((size_t)BLD_ * 4);   // dec state
  float* fb2 = (float*)carve((size_t)BLD_ * 4);   // q / agg / ln tmp
  float* fb3 = (float*)carve((size_t)BLD_ * 4);   // k / attn-out / z
  float* fb4 = (float*)carve((size_t)BLD_ * 4);   // v
  float* fb5 = (float*)carve((size_t)BLD_ * 4);   // seasonal x
  float* fb6 = (float*)carve((size_t)BLD_ * 4);   // trend sum (decoder)
  float* cSeason = (float*)carve((size_t)BLC_ * 4);
  float* cTrend  = (float*)carve((size_t)BLC_ * 4);
  float* cOut    = (float*)carve((size_t)BLC_ * 4);
  float* corr    = (float*)carve((size_t)B_ * L_ * 4);
  float* wts     = (float*)carve((size_t)B_ * TOPK_ * 4);
  int*   delays  = (int*)carve((size_t)B_ * TOPK_ * 4);
  _Float16* xh   = (_Float16*)carve((size_t)BLD_ * 2);
  _Float16* wtA  = (_Float16*)carve((size_t)DFF_ * D_ * 2);
  _Float16* wtB  = (_Float16*)carve((size_t)DFF_ * D_ * 2);
  _Float16* hh   = (_Float16*)carve((size_t)L_ * DFF_ * 2);

  const float* x_enc = P[49];

  // ---- init: decomposition of input + embeddings ----
  k_series_decomp<<<g1(BLC_), 256, 0, stream>>>(x_enc, cSeason, cTrend, C_, BLC_, 0);
  k_circ_conv3<<<g1((long)BLD_), 256, 0, stream>>>(x_enc, P[0], fb0, C_, D_, 0, BLD_);
  k_circ_conv3<<<g1((long)BLD_), 256, 0, stream>>>(cSeason, P[1], fb1, C_, D_, 0, BLD_);

  // ---- encoder: 2 layers ----
  for (int li = 0; li < 2; ++li) {
    int base = 2 + li * 10;
    AttnW aw{P[base], P[base+1], P[base+2], P[base+3], P[base+4], P[base+5], P[base+6], P[base+7]};
    run_attn(stream, fb0, fb0, aw, fb0, fb3, fb2, fb3, fb4, corr, wts, delays, xh, wtA);
    k_series_decomp<<<g1(BLD_), 256, 0, stream>>>(fb3, fb5, nullptr, D_, BLD_, 0);
    run_ffn(stream, fb5, P[base+8], P[base+9], fb3, xh, wtA, wtB, hh);
    k_series_decomp<<<g1(BLD_), 256, 0, stream>>>(fb3, fb0, nullptr, D_, BLD_, 0);
  }
  k_ln_rows<<<g1(BL_), 256, 0, stream>>>(fb0, P[22], P[23], fb2, BL_);
  k_sub_seqmean<<<g1(B_ * D_), 256, 0, stream>>>(fb2, fb0, B_ * D_);

  // ---- decoder: 1 layer ----
  {
    AttnW self {P[24], P[25], P[26], P[27], P[28], P[29], P[30], P[31]};
    AttnW cross{P[32], P[33], P[34], P[35], P[36], P[37], P[38], P[39]};
    run_attn(stream, fb1, fb1, self, fb1, fb3, fb2, fb3, fb4, corr, wts, delays, xh, wtA);
    k_series_decomp<<<g1(BLD_), 256, 0, stream>>>(fb3, fb5, fb6, D_, BLD_, 0);    // t1
    run_attn(stream, fb5, fb0, cross, fb5, fb3, fb2, fb3, fb4, corr, wts, delays, xh, wtA);
    k_series_decomp<<<g1(BLD_), 256, 0, stream>>>(fb3, fb1, fb6, D_, BLD_, 1);    // t2 +=
    run_ffn(stream, fb1, P[40], P[41], fb5, xh, wtA, wtB, hh);
    k_series_decomp<<<g1(BLD_), 256, 0, stream>>>(fb5, fb1, fb6, D_, BLD_, 1);    // t3 +=
    k_circ_conv3<<<g1(BLC_), 256, 0, stream>>>(fb6, P[42], cTrend, D_, C_, 1, BLC_); // trend +=
  }
  k_ln_rows<<<g1(BL_), 256, 0, stream>>>(fb1, P[43], P[44], fb2, BL_);
  k_sub_seqmean<<<g1(B_ * D_), 256, 0, stream>>>(fb2, fb1, B_ * D_);

  // ---- output head ----
  k_proj_c<<<g1(BLC_), 256, 0, stream>>>(fb1, P[45], P[46], cTrend, cOut, BLC_);
  k_pred<<<B_, 256, 0, stream>>>(cOut, P[47], P[48], (float*)d_out);
}
